// transformer_mutis_32452772889196
// MI455X (gfx1250) — matrix operand order repaired and checked by reading, not compiled, not run on hardware
//
#include <hip/hip_runtime.h>
#include <hip/hip_bf16.h>
#include <stdint.h>

typedef __attribute__((ext_vector_type(16))) __bf16 v16bf;
typedef __attribute__((ext_vector_type(8)))  float   v8f;

#define D_MODEL 1024
#define D_FF    2048
#define NHEADS  16
#define DKH     64
#define BATCH   512
#define SEQ     60
#define MROWS   (BATCH*SEQ)   /* 30720 */

// ---------- helpers ----------
__device__ __forceinline__ unsigned short f2b(float f) {
  unsigned u = __builtin_bit_cast(unsigned, f);
  unsigned r = u + 0x7FFFu + ((u >> 16) & 1u);   // round-to-nearest-even
  return (unsigned short)(r >> 16);
}

__device__ __forceinline__ v8f v8zero() {
  v8f z;
  #pragma unroll
  for (int i = 0; i < 8; ++i) z[i] = 0.0f;
  return z;
}

union Frag {            // 32 bytes = one v16bf WMMA operand
  uint4  q[2];
  v16bf  v;
};

// ---------- elementwise ----------
__global__ void cvt_f32_bf16(const float* __restrict__ in,
                             unsigned short* __restrict__ out, long n) {
  long i = (long)blockIdx.x * blockDim.x + threadIdx.x;
  long stride = (long)gridDim.x * blockDim.x;
  for (; i < n; i += stride) out[i] = f2b(in[i]);
}

__global__ void zero_f32(float* p, int n) {
  int i = blockIdx.x * blockDim.x + threadIdx.x;
  if (i < n) p[i] = 0.0f;
}

// ---------- bf16 GEMM: C[M,N] = A[M,K] * B[N,K]^T ----------
// K is a compile-time template parameter (1024 or 2048) so the two-stage
// double-buffered pipeline has literal buffer indices and known trip count.
// MODE 0: store bf16      MODE 1: store fp32 + residual     MODE 2: relu -> bf16
#define BM 128
#define BN 128
#define BK 32
#define KPAD 40   // LDS row stride (elements): 80B, spreads banks, keeps 16B align

template<int MODE, int KDIM>
__global__ __launch_bounds__(256)
void gemm_bf16(const unsigned short* __restrict__ A,
               const unsigned short* __restrict__ Bw,
               int M, int N,
               const float* __restrict__ res,
               float* __restrict__ outF,
               unsigned short* __restrict__ outB) {
  constexpr int T = KDIM / BK;            // 32 or 64 K-tiles (even)
  __shared__ __attribute__((aligned(16))) unsigned short As[2][BM * KPAD];
  __shared__ __attribute__((aligned(16))) unsigned short Bs[2][BN * KPAD];

  const int tid    = threadIdx.x;
  const int lane   = tid & 31;
  const int wid    = tid >> 5;
  const int lane16 = lane & 15;
  const int hf     = lane >> 4;       // half-wave select
  const int waveM  = wid >> 2;        // 0..1
  const int waveN  = wid & 3;         // 0..3

  const int m0 = blockIdx.y * BM;
  const int n0 = blockIdx.x * BN;

  uint4 ra[2], rb[2];
  // per-thread staging: 2 x 16B chunks of the A tile, 2 of the B tile
  auto ld = [&](int t) {
    const int kk = t * BK;
    #pragma unroll
    for (int i = 0; i < 2; ++i) {
      int c = tid + i * 256;                 // chunk id in [0,512)
      int row = c >> 2, col8 = (c & 3) * 8;
      ra[i] = *reinterpret_cast<const uint4*>(A  + (size_t)(m0 + row) * KDIM + kk + col8);
      rb[i] = *reinterpret_cast<const uint4*>(Bw + (size_t)(n0 + row) * KDIM + kk + col8);
    }
  };
  auto st = [&](int buf) {
    #pragma unroll
    for (int i = 0; i < 2; ++i) {
      int c = tid + i * 256;
      int row = c >> 2, col8 = (c & 3) * 8;
      *reinterpret_cast<uint4*>(&As[buf][row * KPAD + col8]) = ra[i];
      *reinterpret_cast<uint4*>(&Bs[buf][row * KPAD + col8]) = rb[i];
    }
  };
  auto pf = [&](int t) {                      // global_prefetch_b8, one row/thread
    int pr = tid & 127;
    const unsigned short* pp =
      (tid < 128 ? A + (size_t)(m0 + pr) * KDIM : Bw + (size_t)(n0 + pr) * KDIM) + t * BK;
    __builtin_prefetch(pp, 0, 1);
  };

  v8f acc[4][2];
  #pragma unroll
  for (int i = 0; i < 4; ++i)
    #pragma unroll
    for (int j = 0; j < 2; ++j) acc[i][j] = v8zero();

  auto comp = [&](int buf) {
    Frag afr[4], bfr[2];
    // A layout: half-wave h holds K = {h*8..h*8+7} U {16+h*8..16+h*8+7}
    #pragma unroll
    for (int mi = 0; mi < 4; ++mi) {
      const unsigned short* p = &As[buf][(waveM * 64 + mi * 16 + lane16) * KPAD + hf * 8];
      afr[mi].q[0] = *reinterpret_cast<const uint4*>(p);
      afr[mi].q[1] = *reinterpret_cast<const uint4*>(p + 16);
    }
    // B layout: half-wave h holds K = {h*8..h*8+7} U {16+h*8..16+h*8+7}
    #pragma unroll
    for (int nj = 0; nj < 2; ++nj) {
      const unsigned short* p = &Bs[buf][(waveN * 32 + nj * 16 + lane16) * KPAD + hf * 8];
      bfr[nj].q[0] = *reinterpret_cast<const uint4*>(p);
      bfr[nj].q[1] = *reinterpret_cast<const uint4*>(p + 16);
    }
    #pragma unroll
    for (int mi = 0; mi < 4; ++mi)
      #pragma unroll
      for (int nj = 0; nj < 2; ++nj)
        acc[mi][nj] = __builtin_amdgcn_wmma_f32_16x16x32_bf16(
            false, afr[mi].v, false, bfr[nj].v, (short)0, acc[mi][nj], false, false);
  };

  // ---- two-stage software pipeline, literal buffer indices ----
  ld(0); st(0); __syncthreads();
  for (int t = 0; t < T - 2; t += 2) {
    // stage A: buf0 holds tile t
    ld(t + 1); pf(t + 2);
    comp(0);
    st(1); __syncthreads();
    // stage B: buf1 holds tile t+1
    ld(t + 2); pf(t + 3);
    comp(1);
    st(0); __syncthreads();
  }
  // tail: buf0 holds tile T-2
  ld(T - 1);
  comp(0);
  st(1); __syncthreads();
  comp(1);

  // epilogue: C 16x16 f32 layout -> row = r + 8*hf, col = lane16
  #pragma unroll
  for (int mi = 0; mi < 4; ++mi) {
    #pragma unroll
    for (int nj = 0; nj < 2; ++nj) {
      int n = n0 + waveN * 32 + nj * 16 + lane16;
      #pragma unroll
      for (int r = 0; r < 8; ++r) {
        int m = m0 + waveM * 64 + mi * 16 + hf * 8 + r;
        size_t idx = (size_t)m * N + n;
        float v = acc[mi][nj][r];
        if (MODE == 0)      outB[idx] = f2b(v);
        else if (MODE == 1) outF[idx] = v + res[idx];
        else                outB[idx] = f2b(v > 0.0f ? v : 0.0f);
      }
    }
  }
}

// ---------- attention: one block (4 waves) per (b,h) ----------
__global__ __launch_bounds__(128)
void attn_kernel(const unsigned short* __restrict__ qb,
                 const unsigned short* __restrict__ kb,
                 const unsigned short* __restrict__ vb,
                 const float* __restrict__ att_sig,
                 unsigned short* __restrict__ ctxb,
                 float* __restrict__ loss_accum) {
  __shared__ __attribute__((aligned(16))) float          sc[64 * 68];
  __shared__ __attribute__((aligned(16))) unsigned short attnb[64 * 72];
  __shared__ __attribute__((aligned(16))) unsigned short vT[64 * 72];
  __shared__ float lred[128];

  const int b = blockIdx.x, h = blockIdx.y;
  const int tid = threadIdx.x;
  const int lane = tid & 31, w = tid >> 5;
  const int lane16 = lane & 15, hf = lane >> 4;
  const size_t headoff = (size_t)h * DKH;
  const uint4 z4 = make_uint4(0, 0, 0, 0);

  // ---- scores = q * k^T / 8 ----
  v8f acc[4];
  #pragma unroll
  for (int i = 0; i < 4; ++i) acc[i] = v8zero();

  const int lq = w * 16 + lane16;
  #pragma unroll
  for (int ks = 0; ks < 2; ++ks) {
    Frag aq;
    if (lq < SEQ) {
      const unsigned short* p =
        qb + (size_t)(b * SEQ + lq) * D_MODEL + headoff + ks * 32 + hf * 8;
      aq.q[0] = *reinterpret_cast<const uint4*>(p);
      aq.q[1] = *reinterpret_cast<const uint4*>(p + 16);
    } else { aq.q[0] = z4; aq.q[1] = z4; }
    #pragma unroll
    for (int nj = 0; nj < 4; ++nj) {
      int lk = nj * 16 + lane16;
      Frag bk;
      if (lk < SEQ) {
        const unsigned short* p =
          kb + (size_t)(b * SEQ + lk) * D_MODEL + headoff + ks * 32 + hf * 8;
        bk.q[0] = *reinterpret_cast<const uint4*>(p);
        bk.q[1] = *reinterpret_cast<const uint4*>(p + 16);
      } else { bk.q[0] = z4; bk.q[1] = z4; }
      acc[nj] = __builtin_amdgcn_wmma_f32_16x16x32_bf16(
          false, aq.v, false, bk.v, (short)0, acc[nj], false, false);
    }
  }
  #pragma unroll
  for (int nj = 0; nj < 4; ++nj)
    #pragma unroll
    for (int r = 0; r < 8; ++r)
      sc[(w * 16 + hf * 8 + r) * 68 + nj * 16 + lane16] = acc[nj][r] * 0.125f;

  // stage v transposed: vT[d][l'] (zero-pad l' >= 60)
  for (int e = tid; e < 64 * 64; e += 128) {
    int d = e >> 6, lp = e & 63;
    vT[d * 72 + lp] =
      (lp < SEQ) ? vb[(size_t)(b * SEQ + lp) * D_MODEL + headoff + d] : (unsigned short)0;
  }
  __syncthreads();

  // ---- softmax + loss (one thread per query row) ----
  float lloss = 0.0f;
  if (tid < SEQ) {
    int row = tid;
    float mx = -1e30f;
    for (int j = 0; j < SEQ; ++j) mx = fmaxf(mx, sc[row * 68 + j]);
    float s = 0.0f;
    for (int j = 0; j < SEQ; ++j) {
      float e = __expf(sc[row * 68 + j] - mx);
      sc[row * 68 + j] = e; s += e;
    }
    float inv = 1.0f / s;
    for (int j = 0; j < SEQ; ++j) {
      float a = sc[row * 68 + j] * inv;
      float d = a - att_sig[row * SEQ + j];
      lloss += d * d;
      attnb[row * 72 + j] = f2b(a);
    }
    for (int j = SEQ; j < 64; ++j) attnb[row * 72 + j] = 0;
  } else if (tid < 64) {
    for (int j = 0; j < 64; ++j) attnb[tid * 72 + j] = 0;
  }
  lred[tid] = lloss;
  __syncthreads();
  for (int s = 64; s > 0; s >>= 1) {
    if (tid < s) lred[tid] += lred[tid + s];
    __syncthreads();
  }
  if (tid == 0) atomicAdd(loss_accum, lred[0]);

  // ---- ctx = attn * v ----
  v8f acc2[4];
  #pragma unroll
  for (int i = 0; i < 4; ++i) acc2[i] = v8zero();
  #pragma unroll
  for (int ks = 0; ks < 2; ++ks) {
    Frag aa;
    {
      const unsigned short* p = &attnb[(w * 16 + lane16) * 72 + ks * 32 + hf * 8];
      aa.q[0] = *reinterpret_cast<const uint4*>(p);
      aa.q[1] = *reinterpret_cast<const uint4*>(p + 16);
    }
    #pragma unroll
    for (int nj = 0; nj < 4; ++nj) {
      const unsigned short* p = &vT[(nj * 16 + lane16) * 72 + ks * 32 + hf * 8];
      Frag bb;
      bb.q[0] = *reinterpret_cast<const uint4*>(p);
      bb.q[1] = *reinterpret_cast<const uint4*>(p + 16);
      acc2[nj] = __builtin_amdgcn_wmma_f32_16x16x32_bf16(
          false, aa.v, false, bb.v, (short)0, acc2[nj], false, false);
    }
  }
  #pragma unroll
  for (int nj = 0; nj < 4; ++nj)
    #pragma unroll
    for (int r = 0; r < 8; ++r) {
      int row = w * 16 + hf * 8 + r;
      if (row < SEQ)
        ctxb[(size_t)(b * SEQ + row) * D_MODEL + headoff + nj * 16 + lane16] =
          f2b(acc2[nj][r]);
    }
}

// ---------- layernorm: one block per row ----------
__global__ __launch_bounds__(256)
void ln_kernel(const float* __restrict__ in, const float* __restrict__ w,
               const float* __restrict__ bia, float* __restrict__ outF,
               unsigned short* __restrict__ outB) {
  __shared__ float r1[256], r2[256];
  int row = blockIdx.x, tid = threadIdx.x;
  const float* rp = in + (size_t)row * D_MODEL;
  float v[4]; float s = 0.0f, sq = 0.0f;
  #pragma unroll
  for (int i = 0; i < 4; ++i) {
    v[i] = rp[tid + i * 256];
    s += v[i]; sq += v[i] * v[i];
  }
  r1[tid] = s; r2[tid] = sq;
  __syncthreads();
  for (int st = 128; st > 0; st >>= 1) {
    if (tid < st) { r1[tid] += r1[tid + st]; r2[tid] += r2[tid + st]; }
    __syncthreads();
  }
  float mean = r1[0] * (1.0f / D_MODEL);
  float var  = r2[0] * (1.0f / D_MODEL) - mean * mean;
  float rstd = rsqrtf(var + 1e-5f);
  #pragma unroll
  for (int i = 0; i < 4; ++i) {
    int c = tid + i * 256;
    float y = (v[i] - mean) * rstd * w[c] + bia[c];
    outF[(size_t)row * D_MODEL + c] = y;
    if (outB) outB[(size_t)row * D_MODEL + c] = f2b(y);
  }
}

// ---------- driver ----------
extern "C" void kernel_launch(void* const* d_in, const int* in_sizes, int n_in,
                              void* d_out, int out_size, void* d_ws, size_t ws_size,
                              hipStream_t stream) {
  const float* x    = (const float*)d_in[0];
  const float* att  = (const float*)d_in[1];
  const float* wq   = (const float*)d_in[2];
  const float* wk   = (const float*)d_in[3];
  const float* wv   = (const float*)d_in[4];
  const float* wo   = (const float*)d_in[5];
  const float* ln1w = (const float*)d_in[6];
  const float* ln1b = (const float*)d_in[7];
  const float* w1   = (const float*)d_in[8];
  const float* w2   = (const float*)d_in[9];
  const float* ln2w = (const float*)d_in[10];
  const float* ln2b = (const float*)d_in[11];

  float* out  = (float*)d_out;
  float* loss = out + (size_t)MROWS * D_MODEL;

  char* ws = (char*)d_ws;
  const size_t MB = 1024 * 1024;
  // weights (bf16), persistent: 16 MB
  unsigned short* wqb = (unsigned short*)(ws + 0 * MB);
  unsigned short* wkb = (unsigned short*)(ws + 2 * MB);
  unsigned short* wvb = (unsigned short*)(ws + 4 * MB);
  unsigned short* wob = (unsigned short*)(ws + 6 * MB);
  unsigned short* w1b = (unsigned short*)(ws + 8 * MB);
  unsigned short* w2b = (unsigned short*)(ws + 12 * MB);
  // 60 MB region reused across lifetimes: xb -> ctxb -> out1b
  unsigned short* xb    = (unsigned short*)(ws + 16 * MB);
  unsigned short* ctxb  = xb;
  unsigned short* out1b = xb;
  // q/k/v (60 MB each); h1b (120 MB) overlays q+k after attention
  unsigned short* qb  = (unsigned short*)(ws + 80 * MB);
  unsigned short* kb  = (unsigned short*)(ws + 140 * MB);
  unsigned short* vbp = (unsigned short*)(ws + 200 * MB);
  unsigned short* h1b = qb;
  float* t0   = (float*)(ws + 260 * MB);   // pre-LN buffer (reused for both LNs)
  float* out1 = (float*)(ws + 380 * MB);   // post-LN1 fp32 (residual for FFN)

  const long nx = (long)MROWS * D_MODEL;
  const long nw = (long)D_MODEL * D_MODEL;
  const long nf = (long)D_FF * D_MODEL;

  cvt_f32_bf16<<<(nx + 255) / 256, 256, 0, stream>>>(x,  xb,  nx);
  cvt_f32_bf16<<<(nw + 255) / 256, 256, 0, stream>>>(wq, wqb, nw);
  cvt_f32_bf16<<<(nw + 255) / 256, 256, 0, stream>>>(wk, wkb, nw);
  cvt_f32_bf16<<<(nw + 255) / 256, 256, 0, stream>>>(wv, wvb, nw);
  cvt_f32_bf16<<<(nw + 255) / 256, 256, 0, stream>>>(wo, wob, nw);
  cvt_f32_bf16<<<(nf + 255) / 256, 256, 0, stream>>>(w1, w1b, nf);
  cvt_f32_bf16<<<(nf + 255) / 256, 256, 0, stream>>>(w2, w2b, nf);
  zero_f32<<<1, 1, 0, stream>>>(loss, 1);

  dim3 g1(D_MODEL / BN, MROWS / BM);   // (8, 240)
  dim3 g2(D_FF / BN,   MROWS / BM);    // (16, 240)

  gemm_bf16<0, D_MODEL><<<g1, 256, 0, stream>>>(xb, wqb, MROWS, D_MODEL, nullptr, nullptr, qb);
  gemm_bf16<0, D_MODEL><<<g1, 256, 0, stream>>>(xb, wkb, MROWS, D_MODEL, nullptr, nullptr, kb);
  gemm_bf16<0, D_MODEL><<<g1, 256, 0, stream>>>(xb, wvb, MROWS, D_MODEL, nullptr, nullptr, vbp);

  attn_kernel<<<dim3(BATCH, NHEADS), 128, 0, stream>>>(qb, kb, vbp, att, ctxb, loss);

  gemm_bf16<1, D_MODEL><<<g1, 256, 0, stream>>>(ctxb, wob, MROWS, D_MODEL, x, t0, nullptr);
  ln_kernel<<<MROWS, 256, 0, stream>>>(t0, ln1w, ln1b, out1, out1b);

  gemm_bf16<2, D_MODEL><<<g2, 256, 0, stream>>>(out1b, w1b, MROWS, D_FF, nullptr, nullptr, h1b);
  gemm_bf16<1, D_FF><<<g1, 256, 0, stream>>>(h1b, w2b, MROWS, D_MODEL, out1, t0, nullptr);
  ln_kernel<<<MROWS, 256, 0, stream>>>(t0, ln2w, ln2b, out, nullptr);

  (void)in_sizes; (void)n_in; (void)out_size; (void)ws_size;
}
